// DecoupledTextDecoder_37881611551166
// MI455X (gfx1250) — matrix operand order repaired and checked by reading, not compiled, not run on hardware
//
#include <hip/hip_runtime.h>
#include <hip/hip_bf16.h>
#include <math.h>

// ---------------------------------------------------------------------------
// DecoupledTextDecoder for MI455X (gfx1250, wave32, WMMA bf16)
// Pipeline:
//   asum   : invAsum[b,t] = 1/sum_hw A[b,t,hw]
//   meta   : steps/used/rowoff/total from textLength (device-side, graph-safe)
//   stage1 : C[(t*128+b), c] = sum_hw feat[b,c,hw] * A[b,t,hw]*invAsum  (WMMA)
//   stage2 : hidden = C @ W^T + bias                                     (WMMA)
//   norm   : norm[m] = ||hidden[m,:]||
//   stage3 : logits = hidden @ protos^T; fused epilogue writes both the
//            gathered `output` (*ALPHA) and `attentionMap` (/ (norm+9e-4))
//   unkcol : writes the UNK_SCR column (index 4000) of both outputs
// Workspace (< 18 MB): invAsum, norm, meta ints, C (8MB), hidden (8MB).
// All B operands are column-major in memory, so WMMA fragments are loaded
// straight from global as contiguous 64B runs per lane (no LDS shuffle).
// Out-of-range proto columns are handled with a branchless index clamp
// (clamped columns produce garbage D columns that are never stored), so the
// k-loops contain no EXEC divergence around the WMMAs.
// ---------------------------------------------------------------------------

typedef __attribute__((ext_vector_type(16))) __bf16 v16bf;
typedef __attribute__((ext_vector_type(8)))  float  v8f;

#define NB 128
#define NC 512
#define NHW 1024
#define NT 32
#define NP 4000
#define NCLASS 4001
#define NROWS (NT * NB)   // 4096

// workspace byte offsets
#define WS_INVASUM 0                       // 4096 f32
#define WS_NORM    (16 * 1024)             // 4096 f32
#define WS_META    (32 * 1024)             // ints: [0]=steps [1]=total used[128]@2 rowoff[128]@130
#define WS_C       (1 * 1024 * 1024)       // 4096x512 f32 (8 MB)
#define WS_HIDDEN  (WS_C + NROWS * NC * 4) // 8 MB

// ---- WMMA fragment loaders (direct from global, contiguous per lane) ------
// A fragment 16(M)x32(K) bf16: lane L -> M = L&15, Kbase = (L>>4)*16,
//   elements 0..15 = K  Kbase..Kbase+15
__device__ __forceinline__ v16bf load_a_frag(const float* __restrict__ A,
                                             long lda, int k0, int lane) {
  const int M  = lane & 15;
  const int Kb = (lane >> 4) << 4;
  const float* p = A + (long)M * lda + k0 + Kb;
  v16bf r;
#pragma unroll
  for (int e = 0; e < 8; ++e) r[e] = (__bf16)p[e];
#pragma unroll
  for (int e = 0; e < 8; ++e) r[8 + e] = (__bf16)p[8 + e];
  return r;
}

__device__ __forceinline__ v16bf load_a_frag_scaled(const float* __restrict__ A,
                                                    long lda, int k0, float scale,
                                                    int lane) {
  const int M  = lane & 15;
  const int Kb = (lane >> 4) << 4;
  const float* p = A + (long)M * lda + k0 + Kb;
  v16bf r;
#pragma unroll
  for (int e = 0; e < 8; ++e) r[e] = (__bf16)(p[e] * scale);
#pragma unroll
  for (int e = 0; e < 8; ++e) r[8 + e] = (__bf16)(p[8 + e] * scale);
  return r;
}

// B fragment 32(K)x16(N) bf16 from a column-major operand
// (B[k][n] = Bcm[n*ldb + k]): lane L -> N = n0 + (L&15), holds 16 consecutive
// K starting at (L>>4)*16. N is clamped branchlessly into [0, nmax);
// clamped lanes produce duplicate (discarded) output columns.
__device__ __forceinline__ v16bf load_b_frag(const float* __restrict__ Bcm,
                                             long ldb, int n0, int k0,
                                             int nmax, int lane) {
  int N = n0 + (lane & 15);
  N = (N < nmax) ? N : (nmax - 1);        // v_min, no EXEC divergence
  const int Kb = (lane >> 4) << 4;
  const float* p = Bcm + (long)N * ldb + k0 + Kb;
  v16bf r;
#pragma unroll
  for (int e = 0; e < 16; ++e) r[e] = (__bf16)p[e];
  return r;
}

// ---- reductions ------------------------------------------------------------
__global__ __launch_bounds__(256) void asum_kernel(const float* __restrict__ A,
                                                   float* __restrict__ invAsum) {
  __shared__ float sb[256];
  const int bt = blockIdx.x;                      // 0..4095 = b*32+t
  const float* p = A + (long)bt * NHW;
  float s = 0.f;
  for (int i = threadIdx.x; i < NHW; i += 256) s += p[i];
  sb[threadIdx.x] = s;
  __syncthreads();
  for (int w = 128; w > 0; w >>= 1) {
    if (threadIdx.x < w) sb[threadIdx.x] += sb[threadIdx.x + w];
    __syncthreads();
  }
  if (threadIdx.x == 0) invAsum[bt] = 1.0f / sb[0];
}

__global__ __launch_bounds__(256) void norm_kernel(const float* __restrict__ H,
                                                   float* __restrict__ nrm) {
  __shared__ float sb[256];
  const int m = blockIdx.x;                       // 0..4095
  const float* p = H + (long)m * NC;
  float s = 0.f;
  for (int i = threadIdx.x; i < NC; i += 256) { float v = p[i]; s += v * v; }
  sb[threadIdx.x] = s;
  __syncthreads();
  for (int w = 128; w > 0; w >>= 1) {
    if (threadIdx.x < w) sb[threadIdx.x] += sb[threadIdx.x + w];
    __syncthreads();
  }
  if (threadIdx.x == 0) nrm[m] = sqrtf(sb[0]);
}

__global__ void meta_kernel(const long long* __restrict__ textLength,
                            int* __restrict__ meta) {
  if (threadIdx.x == 0 && blockIdx.x == 0) {
    int steps = 0, total = 0;
    for (int b = 0; b < NB; ++b) {
      int u = (int)textLength[b];
      if (u > NT) u = NT;
      meta[2 + b]   = u;        // used[b]
      meta[130 + b] = total;    // exclusive prefix (row offset)
      total += u;
      if (u > steps) steps = u;
    }
    meta[0] = steps;
    meta[1] = total;
  }
}

// ---- stage 1: attention pooling GEMM --------------------------------------
__global__ __launch_bounds__(32)
void stage1_pool(const float* __restrict__ feat, const float* __restrict__ A,
                 const float* __restrict__ invAsum, float* __restrict__ C) {
  const int b    = blockIdx.x;            // batch
  const int n0   = blockIdx.y * 64;       // c tile (8 tiles of 64)
  const int m0   = blockIdx.z * 16;       // t tile (2 tiles of 16)
  const int lane = threadIdx.x;

  const float* Ab = A + ((long)b * NT + m0) * NHW;    // rows = t, lda = 1024
  const float* Fb = feat + (long)b * NC * NHW;        // B col-major, ldb = 1024
  const float scale = invAsum[b * NT + m0 + (lane & 15)];

  v8f acc0 = {}, acc1 = {}, acc2 = {}, acc3 = {};
  for (int k0 = 0; k0 < NHW; k0 += 32) {
    v16bf a  = load_a_frag_scaled(Ab, NHW, k0, scale, lane);
    v16bf b0 = load_b_frag(Fb, NHW, n0 +  0, k0, NC, lane);
    v16bf b1 = load_b_frag(Fb, NHW, n0 + 16, k0, NC, lane);
    v16bf b2 = load_b_frag(Fb, NHW, n0 + 32, k0, NC, lane);
    v16bf b3 = load_b_frag(Fb, NHW, n0 + 48, k0, NC, lane);
    acc0 = __builtin_amdgcn_wmma_f32_16x16x32_bf16(false, a, false, b0, (short)0, acc0, false, false);
    acc1 = __builtin_amdgcn_wmma_f32_16x16x32_bf16(false, a, false, b1, (short)0, acc1, false, false);
    acc2 = __builtin_amdgcn_wmma_f32_16x16x32_bf16(false, a, false, b2, (short)0, acc2, false, false);
    acc3 = __builtin_amdgcn_wmma_f32_16x16x32_bf16(false, a, false, b3, (short)0, acc3, false, false);
  }
  const int Mofs = (lane >> 4) << 3;      // 0 or 8
  const int n    = lane & 15;
#pragma unroll
  for (int r = 0; r < 8; ++r) {
    const int t = m0 + r + Mofs;
    float* dst = C + ((long)t * NB + b) * NC + n0 + n;
    dst[0]  = acc0[r];
    dst[16] = acc1[r];
    dst[32] = acc2[r];
    dst[48] = acc3[r];
  }
}

// ---- stage 2: hidden = C @ W^T + bias -------------------------------------
__global__ __launch_bounds__(32)
void stage2_hidden(const float* __restrict__ C, const float* __restrict__ W,
                   const float* __restrict__ bias, float* __restrict__ H) {
  const int m0   = blockIdx.x * 16;       // 256 row tiles
  const int n0   = blockIdx.y * 64;       // 8 col tiles
  const int lane = threadIdx.x;
  const float* Am = C + (long)m0 * NC;

  v8f acc0 = {}, acc1 = {}, acc2 = {}, acc3 = {};
  for (int k0 = 0; k0 < NC; k0 += 32) {
    v16bf a  = load_a_frag(Am, NC, k0, lane);
    v16bf b0 = load_b_frag(W, NC, n0 +  0, k0, NC, lane);
    v16bf b1 = load_b_frag(W, NC, n0 + 16, k0, NC, lane);
    v16bf b2 = load_b_frag(W, NC, n0 + 32, k0, NC, lane);
    v16bf b3 = load_b_frag(W, NC, n0 + 48, k0, NC, lane);
    acc0 = __builtin_amdgcn_wmma_f32_16x16x32_bf16(false, a, false, b0, (short)0, acc0, false, false);
    acc1 = __builtin_amdgcn_wmma_f32_16x16x32_bf16(false, a, false, b1, (short)0, acc1, false, false);
    acc2 = __builtin_amdgcn_wmma_f32_16x16x32_bf16(false, a, false, b2, (short)0, acc2, false, false);
    acc3 = __builtin_amdgcn_wmma_f32_16x16x32_bf16(false, a, false, b3, (short)0, acc3, false, false);
  }
  const int Mofs = (lane >> 4) << 3;
  const int n    = n0 + (lane & 15);
#pragma unroll
  for (int r = 0; r < 8; ++r) {
    float* dst = H + (long)(m0 + r + Mofs) * NC + n;
    dst[0]  = acc0[r] + bias[n];
    dst[16] = acc1[r] + bias[n + 16];
    dst[32] = acc2[r] + bias[n + 32];
    dst[48] = acc3[r] + bias[n + 48];
  }
}

// ---- stage 3: logits = hidden @ protos^T, fused output epilogue -----------
__global__ __launch_bounds__(32)
void stage3_logits(const float* __restrict__ H, const float* __restrict__ P,
                   const float* __restrict__ nrm, const int* __restrict__ meta,
                   const float* __restrict__ ALPHA, float* __restrict__ out) {
  const int m0   = blockIdx.x * 16;       // 256 row tiles over 4096 rows
  const int n0   = blockIdx.y * 64;       // 63 col tiles over 4000 (clamped)
  const int lane = threadIdx.x;

  const float* Am = H + (long)m0 * NC;
  v8f acc[4] = {v8f{}, v8f{}, v8f{}, v8f{}};
  for (int k0 = 0; k0 < NC; k0 += 32) {
    v16bf a  = load_a_frag(Am, NC, k0, lane);
    v16bf b0 = load_b_frag(P, NC, n0 +  0, k0, NP, lane);
    v16bf b1 = load_b_frag(P, NC, n0 + 16, k0, NP, lane);
    v16bf b2 = load_b_frag(P, NC, n0 + 32, k0, NP, lane);
    v16bf b3 = load_b_frag(P, NC, n0 + 48, k0, NP, lane);
    acc[0] = __builtin_amdgcn_wmma_f32_16x16x32_bf16(false, a, false, b0, (short)0, acc[0], false, false);
    acc[1] = __builtin_amdgcn_wmma_f32_16x16x32_bf16(false, a, false, b1, (short)0, acc[1], false, false);
    acc[2] = __builtin_amdgcn_wmma_f32_16x16x32_bf16(false, a, false, b2, (short)0, acc[2], false, false);
    acc[3] = __builtin_amdgcn_wmma_f32_16x16x32_bf16(false, a, false, b3, (short)0, acc[3], false, false);
  }

  const int   steps  = meta[0];
  const int   total  = meta[1];
  const int*  used   = meta + 2;
  const int*  rowoff = meta + 130;
  const float alpha  = ALPHA[0];
  const size_t attBase = (size_t)total * NCLASS;

  const int Mofs = (lane >> 4) << 3;
  const int nn   = lane & 15;
#pragma unroll
  for (int g = 0; g < 4; ++g) {
    const int n = n0 + g * 16 + nn;
    if (n >= NP) continue;               // post-WMMA, store guard only
#pragma unroll
    for (int r = 0; r < 8; ++r) {
      const int m = m0 + r + Mofs;       // row = t*128 + b
      const int t = m >> 7;
      const int b = m & 127;
      const float v = acc[g][r];
      if (t < steps)
        out[attBase + (size_t)m * NCLASS + n] = v / (nrm[m] + 0.0009f);
      if (t < used[b])
        out[(size_t)(rowoff[b] + t) * NCLASS + n] = v * alpha;
    }
  }
}

// ---- UNK column (index 4000) of both outputs ------------------------------
__global__ __launch_bounds__(256)
void unkcol_kernel(const int* __restrict__ meta, const float* __restrict__ UNK,
                   float* __restrict__ out) {
  const int idx = blockIdx.x * 256 + threadIdx.x;   // 0..4095 = t*128+b
  if (idx >= NROWS) return;
  const int steps = meta[0];
  const int total = meta[1];
  const int* used = meta + 2;
  const int* rowoff = meta + 130;
  const int t = idx >> 7, b = idx & 127;
  const float u = UNK[0];
  const size_t attBase = (size_t)total * NCLASS;
  if (t < steps) out[attBase + (size_t)idx * NCLASS + NP] = u;
  if (t < used[b]) out[(size_t)(rowoff[b] + t) * NCLASS + NP] = u;
}

// ---------------------------------------------------------------------------
extern "C" void kernel_launch(void* const* d_in, const int* in_sizes, int n_in,
                              void* d_out, int out_size, void* d_ws, size_t ws_size,
                              hipStream_t stream) {
  const float*     feature = (const float*)d_in[0];
  const float*     A       = (const float*)d_in[1];
  const float*     protos  = (const float*)d_in[2];
  const float*     W       = (const float*)d_in[3];
  const float*     bias    = (const float*)d_in[4];
  const float*     UNK     = (const float*)d_in[5];
  const float*     ALPHA   = (const float*)d_in[6];
  // d_in[7] = labels (arange -> identity scatter, unused)
  const long long* textLen = (const long long*)d_in[8];
  float* out = (float*)d_out;

  char* ws = (char*)d_ws;
  float* invAsum = (float*)(ws + WS_INVASUM);
  float* nrm     = (float*)(ws + WS_NORM);
  int*   meta    = (int*)  (ws + WS_META);
  float* C       = (float*)(ws + WS_C);
  float* H       = (float*)(ws + WS_HIDDEN);

  asum_kernel<<<NROWS, 256, 0, stream>>>(A, invAsum);
  meta_kernel<<<1, 32, 0, stream>>>(textLen, meta);
  stage1_pool<<<dim3(NB, NC / 64, NT / 16), 32, 0, stream>>>(feature, A, invAsum, C);
  stage2_hidden<<<dim3(NROWS / 16, NC / 64), 32, 0, stream>>>(C, W, bias, H);
  norm_kernel<<<NROWS, 256, 0, stream>>>(H, nrm);
  stage3_logits<<<dim3(NROWS / 16, (NP + 63) / 64), 32, 0, stream>>>(H, protos, nrm, meta, ALPHA, out);
  unkcol_kernel<<<(NROWS + 255) / 256, 256, 0, stream>>>(meta, UNK, out);
}
